// FirstFFTLinearLayer_34179349741979
// MI455X (gfx1250) — matrix operand order repaired and checked by reading, not compiled, not run on hardware
//
#include <hip/hip_runtime.h>

// Problem constants (fixed by the reference): B=16, N=64, H=W=256.
#define NF     64          // FFT length (complex outputs per pixel)
#define HW     65536       // H*W
#define NHW    (NF * HW)   // per-batch plane stride in elements
#define K2     128         // 2*NF : GEMM K and N dimensions
#define TILE_P 64          // pixels per tile
#define TILES  8           // tiles per (persistent) workgroup
#define LDA    136         // f16 row stride for A / weight tiles (pad vs 128)
#define LDO    132         // f32 row stride for output staging (pad vs 128)

typedef __attribute__((ext_vector_type(16))) _Float16 v16h;
typedef __attribute__((ext_vector_type(8)))  _Float16 v8h;
typedef __attribute__((ext_vector_type(8)))  float    v8f;
typedef __attribute__((ext_vector_type(4)))  float    v4f;  // clang vector: OK for NT builtins
typedef __attribute__((ext_vector_type(2)))  float    v2f;

// Weight tile is persistent across the whole workgroup lifetime; only the
// input staging (A) and output staging (O) buffers overlap (phase-separated
// by barriers).
struct SMem {
  _Float16 Wt[K2 * LDA];               // 34816 B, persistent
  union {
    _Float16 A[TILE_P * LDA];          // 17408 B
    float    O[TILE_P * LDO];          // 33792 B
  } u;
};

__device__ __forceinline__ v16h cat16(v8h lo, v8h hi) {
  return __builtin_shufflevector(lo, hi, 0,1,2,3,4,5,6,7,8,9,10,11,12,13,14,15);
}

__global__ __launch_bounds__(256, 2)
void fft_linear_wmma(const float* __restrict__ real,
                     const float* __restrict__ imag,
                     const float* __restrict__ wgt,
                     float* __restrict__ outf) {
  __shared__ SMem sm;
  _Float16* sW = sm.Wt;
  _Float16* sA = sm.u.A;
  float*    sO = sm.u.O;

  const int tid      = threadIdx.x;
  const int lane     = tid & 31;
  const int wave     = tid >> 5;
  const int half_sel = lane >> 4;   // wave32 half: selects K sub-block
  const int mrow     = lane & 15;   // matrix row/col within 16

  // Persistent WG: 512 consecutive pixels (8 tiles of 64), single b-plane.
  const size_t pwg = (size_t)blockIdx.x * (TILE_P * TILES);
  const int    bb  = (int)(pwg / HW);
  const int    pin = (int)(pwg % HW);
  const float* rbase = real + (size_t)bb * NHW + pin;
  const float* ibase = imag + (size_t)bb * NHW + pin;
  float*       ob    = outf + 2ull * ((size_t)bb * NHW + pin);

  // ---- Stage weight [j][k] as f16 once per workgroup (regular-temporal:
  // this is the reuse case and lives in L2 across the dispatch).
  #pragma unroll
  for (int it = 0; it < (K2 * K2 / 4) / 256; ++it) {       // 16 iters
    int idx = it * 256 + tid;
    int j   = idx >> 5;              // 0..127
    int k4  = (idx & 31) << 2;       // 0..124
    v4f v = *(const v4f*)(wgt + j * K2 + k4);
    _Float16* d = sW + j * LDA + k4;
    d[0] = (_Float16)v.x; d[1] = (_Float16)v.y;
    d[2] = (_Float16)v.z; d[3] = (_Float16)v.w;
  }
  __syncthreads();

  // ---- Hoist B fragments (wave's 16 output columns x full K=128) into
  // registers for the entire tile loop. B in contiguous k order (same as A): lane = col,
  // lanes 0-15 hold K kb..kb+15, lanes 16-31 hold K kb+16..kb+31.
  const int j0 = wave * 16;
  v16h bfrag[4];
  #pragma unroll
  for (int kc = 0; kc < 4; ++kc) {
    const _Float16* rowB = sW + (j0 + mrow) * LDA + kc * 32 + half_sel * 16;
    bfrag[kc] = cat16(*(const v8h*)rowB, *(const v8h*)(rowB + 8));
  }

  for (int t = 0; t < TILES; ++t) {
    const int tp = t * TILE_P;       // tile pixel offset within WG range

    // ---- Stage input tile A[pixel][k] (k: 0..63 real, 64..127 imag).
    // Pixels contiguous per k-plane -> coalesced NT float4 loads (streamed
    // once; keep them out of L2's way).
    #pragma unroll
    for (int it = 0; it < (TILE_P * K2 / 4) / 256; ++it) { // 8 iters
      int idx = it * 256 + tid;
      int k   = idx >> 4;            // 0..127
      int p4  = (idx & 15) << 2;     // 0..60
      const float* src = (k < NF)
          ? (rbase + (size_t)k * HW + tp + p4)
          : (ibase + (size_t)(k - NF) * HW + tp + p4);
      v4f v = __builtin_nontemporal_load((const v4f*)src);
      sA[(p4 + 0) * LDA + k] = (_Float16)v.x;
      sA[(p4 + 1) * LDA + k] = (_Float16)v.y;
      sA[(p4 + 2) * LDA + k] = (_Float16)v.z;
      sA[(p4 + 3) * LDA + k] = (_Float16)v.w;
    }
    __syncthreads();

    // ---- WMMA: wave owns 16 j-columns, 4 pixel subtiles, K=128 in 4 steps.
    v8f acc[TILE_P / 16] = {};
    #pragma unroll
    for (int kc = 0; kc < 4; ++kc) {
      #pragma unroll
      for (int pt = 0; pt < TILE_P / 16; ++pt) {
        // A in contiguous k order (same as B): lane = pixel row; halves {kb..kb+15}
        // (lanes 16-31 shifted by +16).
        const _Float16* rowA =
            sA + (pt * 16 + mrow) * LDA + kc * 32 + half_sel * 16;
        v16h afrag = cat16(*(const v8h*)rowA, *(const v8h*)(rowA + 8));
        acc[pt] = __builtin_amdgcn_wmma_f32_16x16x32_f16(
            false, afrag, false, bfrag[kc], (short)0, acc[pt], false, false);
      }
    }
    __syncthreads();   // all A reads complete before O overwrites the union

    // ---- Spill D tiles: D VGPR r -> row (r or 8+r), lane -> column.
    #pragma unroll
    for (int pt = 0; pt < TILE_P / 16; ++pt) {
      int prow = pt * 16 + half_sel * 8;
      int jc   = j0 + mrow;
      #pragma unroll
      for (int r = 0; r < 8; ++r)
        sO[(prow + r) * LDO + jc] = acc[pt][r];
    }
    __syncthreads();

    // ---- Coalesced complex64 NT store: (re, im) = (col j, col j+64),
    // pixels contiguous along the store address -> 256B/wave bursts.
    #pragma unroll
    for (int it = 0; it < (TILE_P * NF) / 256; ++it) {     // 16 iters
      int idx = it * 256 + tid;
      int j = idx >> 6;              // 0..63
      int p = idx & 63;              // 0..63
      v2f v = {sO[p * LDO + j], sO[p * LDO + NF + j]};
      __builtin_nontemporal_store(
          v, (v2f*)(ob + 2ull * ((size_t)j * HW + tp + p)));
    }
    __syncthreads();   // O reads complete before next tile's A staging
  }
}

extern "C" void kernel_launch(void* const* d_in, const int* in_sizes, int n_in,
                              void* d_out, int out_size, void* d_ws, size_t ws_size,
                              hipStream_t stream) {
  const float* real = (const float*)d_in[0];   // [16, 64, 256, 256] f32
  const float* imag = (const float*)d_in[1];   // [16, 64, 256, 256] f32
  const float* wgt  = (const float*)d_in[2];   // [128, 128] f32
  float* outf = (float*)d_out;                 // [16, 64, 256, 256] complex64

  const int M = 16 * HW;                       // B*H*W pixels
  dim3 grid(M / (TILE_P * TILES)), block(256); // 2048 persistent WGs
  hipLaunchKernelGGL(fft_linear_wmma, grid, block, 0, stream,
                     real, imag, wgt, outf);
}
